// Model_55224689492095
// MI455X (gfx1250) — matrix operand order repaired and checked by reading, not compiled, not run on hardware
//
#include <hip/hip_runtime.h>
#include <hip/hip_bf16.h>

// ---------------- types ----------------
typedef __attribute__((ext_vector_type(16))) __bf16 v16bf;
typedef __attribute__((ext_vector_type(8)))  float  v8f;
typedef __attribute__((ext_vector_type(2)))  float  v2f;

// ---------------- device helpers ----------------
__device__ __forceinline__ float gelu_tanh(float x) {
  float x3 = x * x * x;
  return 0.5f * x * (1.0f + tanhf(0.7978845608028654f * (x + 0.044715f * x3)));
}

// ---------------- generic bf16 WMMA GEMM ----------------
// Block = 4 waves; each wave computes a 16(M) x 64(N) strip with 4 WMMA
// accumulators, sharing one A fragment per 32-K chunk. Block tile = 64x64.
// C[z][m][n] = act( alpha*sum_k A[m][k]*B[k][n] + bias[n] + res[(m&rmask)][n] )
// A must have unit K stride.
// B fast path: b_sn==1 (row-major weights)  -> direct per-element global loads.
// B slow path: b_sn!=1 (requires b_sk==1, i.e. K^T view) -> async-to-LDS
//              cooperative staging of the 64(N) x 32(K) tile, then LDS reads.
__global__ void __launch_bounds__(128)
gemm_bf16(const float* __restrict__ A, long long a_sm,
          long long a_sb, long long a_sh,
          const float* __restrict__ B, long long b_sk, long long b_sn,
          long long b_sb, long long b_sh,
          float* __restrict__ C, long long c_sm, long long c_sb, long long c_sh,
          const float* __restrict__ bias,
          const float* __restrict__ res, long long r_sm, int rmask,
          int M, int N, int K, int nh, float alpha, int act) {
  __shared__ float tileB[64 * 33];   // [n(64)][k(32)] with pad, slow path only
  int lane = threadIdx.x;
  int wv   = threadIdx.y;
  int m0 = (blockIdx.y * 4 + wv) * 16;
  int nb = blockIdx.x * 64;
  if (m0 >= M || nb >= N) return;
  int z  = blockIdx.z;
  int bb = z / nh, hh = z % nh;
  const float* Ab = A + (long long)bb * a_sb + (long long)hh * a_sh;
  const float* Bb = B + (long long)bb * b_sb + (long long)hh * b_sh;
  float*       Cb = C + (long long)bb * c_sb + (long long)hh * c_sh;

  int half = lane >> 4;          // 0|1
  int l15  = lane & 15;
  int tid  = wv * 32 + lane;     // 0..127
  const float* ap = Ab + (long long)(m0 + l15) * a_sm;  // this lane's A row

  v8f acc0 = {}, acc1 = {}, acc2 = {}, acc3 = {};
  for (int k0 = 0; k0 < K; k0 += 32) {
    // ---- A fragment: lane -> M row; elems 0..7 = K run [k0+8h, +8),
    //      elems 8..15 = K run [k0+16+8h, +8)  (ISA 16-bit A layout) ----
    const float4* r0 = (const float4*)(ap + k0 + (half << 3));
    const float4* r1 = (const float4*)(ap + k0 + 16 + (half << 3));
    float4 q0 = r0[0], q1 = r0[1], q2 = r1[0], q3 = r1[1];
    __builtin_prefetch(ap + k0 + 64, 0, 3);   // A chunk two ahead -> global_prefetch
    v16bf a;
    a[0] = (__bf16)q0.x; a[1] = (__bf16)q0.y; a[2]  = (__bf16)q0.z; a[3]  = (__bf16)q0.w;
    a[4] = (__bf16)q1.x; a[5] = (__bf16)q1.y; a[6]  = (__bf16)q1.z; a[7]  = (__bf16)q1.w;
    a[8] = (__bf16)q2.x; a[9] = (__bf16)q2.y; a[10] = (__bf16)q2.z; a[11] = (__bf16)q2.w;
    a[12] = (__bf16)q3.x; a[13] = (__bf16)q3.y; a[14] = (__bf16)q3.z; a[15] = (__bf16)q3.w;

    if (b_sn == 1) {
      // ---- fast path: B rows contiguous in N; lane -> N (l15), elem -> K (table order) ----
      const float* bk = Bb + nb + l15;
#pragma unroll
      for (int t = 0; t < 4; ++t) {
        const float* bp = bk + t * 16;
        v16bf b;
#pragma unroll
        for (int e = 0; e < 16; ++e) {
          int kk = k0 + (e < 8 ? (half << 3) + e : 16 + (half << 3) + (e - 8));
          b[e] = (__bf16)bp[(long long)kk * b_sk];
        }
        if (t == 0) acc0 = __builtin_amdgcn_wmma_f32_16x16x32_bf16(false, a, false, b, (short)0, acc0, false, false);
        if (t == 1) acc1 = __builtin_amdgcn_wmma_f32_16x16x32_bf16(false, a, false, b, (short)0, acc1, false, false);
        if (t == 2) acc2 = __builtin_amdgcn_wmma_f32_16x16x32_bf16(false, a, false, b, (short)0, acc2, false, false);
        if (t == 3) acc3 = __builtin_amdgcn_wmma_f32_16x16x32_bf16(false, a, false, b, (short)0, acc3, false, false);
      }
    } else {
      // ---- slow path (b_sk==1): cooperatively stage B tile [64n x 32k]
      //      into LDS via async copies, then read fragments from LDS ----
      __syncthreads();   // previous iteration's LDS reads complete
#pragma unroll
      for (int i = 0; i < 4; ++i) {
        int p    = tid + i * 128;       // 0..511
        int key  = p >> 3;              // 0..63   (N within tile)
        int quad = p & 7;               // 0..7    (K quad)
        const float* src = Bb + (long long)(nb + key) * b_sn + k0 + quad * 4;
        unsigned dst = (unsigned)(size_t)&tileB[key * 33 + quad * 4];
        asm volatile("global_load_async_to_lds_b128 %0, %1, off"
                     :: "v"(dst), "v"(src) : "memory");
      }
      asm volatile("s_wait_asynccnt 0" ::: "memory");
      __syncthreads();   // all waves' async stages visible
#pragma unroll
      for (int t = 0; t < 4; ++t) {
        const float* bp = &tileB[(t * 16 + l15) * 33];  // lane -> n, elem -> k (table order)
        v16bf b;
#pragma unroll
        for (int e = 0; e < 16; ++e) b[e] = (__bf16)bp[e < 8 ? (half << 3) + e : 16 + (half << 3) + (e - 8)];
        if (t == 0) acc0 = __builtin_amdgcn_wmma_f32_16x16x32_bf16(false, a, false, b, (short)0, acc0, false, false);
        if (t == 1) acc1 = __builtin_amdgcn_wmma_f32_16x16x32_bf16(false, a, false, b, (short)0, acc1, false, false);
        if (t == 2) acc2 = __builtin_amdgcn_wmma_f32_16x16x32_bf16(false, a, false, b, (short)0, acc2, false, false);
        if (t == 3) acc3 = __builtin_amdgcn_wmma_f32_16x16x32_bf16(false, a, false, b, (short)0, acc3, false, false);
      }
    }
  }

  // ---- epilogue: C/D layout lane -> N (lane&15), VGPR r -> M = r + 8*half ----
#pragma unroll
  for (int t = 0; t < 4; ++t) {
    v8f acc = (t == 0) ? acc0 : (t == 1) ? acc1 : (t == 2) ? acc2 : acc3;
    int n = nb + t * 16 + l15;
    float bv = bias ? bias[n] : 0.0f;
#pragma unroll
    for (int r = 0; r < 8; ++r) {
      int m = m0 + r + (half << 3);
      float v = acc[r] * alpha + bv;
      if (res) v += res[(long long)(m & rmask) * r_sm + n];
      if (act == 1) v = gelu_tanh(v);
      Cb[(long long)m * c_sm + n] = v;
    }
  }
}

// ---------------- LayerNorm (one block of 256 per row) ----------------
__global__ void __launch_bounds__(256)
layernorm_k(const float* __restrict__ x, const float* __restrict__ g,
            const float* __restrict__ b, float* __restrict__ y, int D) {
  __shared__ float red[256];
  int row = blockIdx.x, t = threadIdx.x;
  const float* xr = x + (long long)row * D;
  float s = 0.f;
  for (int i = t; i < D; i += 256) s += xr[i];
  red[t] = s; __syncthreads();
  for (int o = 128; o > 0; o >>= 1) { if (t < o) red[t] += red[t + o]; __syncthreads(); }
  float mean = red[0] / (float)D; __syncthreads();
  float s2 = 0.f;
  for (int i = t; i < D; i += 256) { float d = xr[i] - mean; s2 += d * d; }
  red[t] = s2; __syncthreads();
  for (int o = 128; o > 0; o >>= 1) { if (t < o) red[t] += red[t + o]; __syncthreads(); }
  float inv = rsqrtf(red[0] / (float)D + 1e-5f);
  float* yr = y + (long long)row * D;
  for (int i = t; i < D; i += 256) yr[i] = (xr[i] - mean) * inv * g[i] + b[i];
}

// ---------------- softmax over rows of 256 (1 wave per row) ----------------
__global__ void __launch_bounds__(256)
softmax256(float* __restrict__ s, int rows) {
  int r = blockIdx.x * blockDim.y + threadIdx.y;
  if (r >= rows) return;
  int lane = threadIdx.x;
  float* p = s + (long long)r * 256;
  float v[8]; float mx = -3.4e38f;
#pragma unroll
  for (int e = 0; e < 8; ++e) { v[e] = p[e * 32 + lane]; mx = fmaxf(mx, v[e]); }
  for (int o = 1; o < 32; o <<= 1) mx = fmaxf(mx, __shfl_xor(mx, o, 32));
  float sum = 0.f;
#pragma unroll
  for (int e = 0; e < 8; ++e) { v[e] = expf(v[e] - mx); sum += v[e]; }
  for (int o = 1; o < 32; o <<= 1) sum += __shfl_xor(sum, o, 32);
  float inv = 1.0f / sum;
#pragma unroll
  for (int e = 0; e < 8; ++e) p[e * 32 + lane] = v[e] * inv;
}

// ---------------- row squared-norm (rows of 256) ----------------
__global__ void __launch_bounds__(256)
rownorm256(const float* __restrict__ x, float* __restrict__ out) {
  __shared__ float red[256];
  int row = blockIdx.x, t = threadIdx.x;
  float v = x[(long long)row * 256 + t];
  red[t] = v * v; __syncthreads();
  for (int o = 128; o > 0; o >>= 1) { if (t < o) red[t] += red[t + o]; __syncthreads(); }
  if (t == 0) out[row] = red[0];
}

// ---------------- VQ argmin via f32 WMMA (16x16x4) ----------------
__global__ void __launch_bounds__(128)
vq_argmin(const float* __restrict__ Z, const float* __restrict__ CB,
          const float* __restrict__ zn, const float* __restrict__ cn,
          unsigned long long* __restrict__ best, int Dc) {
  int lane = threadIdx.x;
  int n0 = (blockIdx.x * blockDim.y + threadIdx.y) * 16;  // code tile
  int m0 = blockIdx.y * 16;                               // token tile
  int half = lane >> 4, l15 = lane & 15;
  const float* zrow = Z  + (long long)(m0 + l15) * Dc + (half << 1);
  const float* crow = CB + (long long)(n0 + l15) * Dc + (half << 1);
  v8f acc = {};
  for (int k0 = 0; k0 < Dc; k0 += 4) {
    v2f a = *(const v2f*)(zrow + k0);
    v2f b = *(const v2f*)(crow + k0);
    acc = __builtin_amdgcn_wmma_f32_16x16x4_f32(false, a, false, b,
                                                (short)0, acc, false, false);
  }
  int n = n0 + l15;
  float cnv = cn[n];
#pragma unroll
  for (int r = 0; r < 8; ++r) {
    int m = m0 + r + (half << 3);
    float d2 = fmaxf(zn[m] + cnv - 2.0f * acc[r], 0.0f);
    unsigned long long p =
        (((unsigned long long)__float_as_uint(d2)) << 32) | (unsigned)n;
#pragma unroll
    for (int o = 1; o < 16; o <<= 1) {
      unsigned long long q = __shfl_xor(p, o, 32);
      p = (q < p) ? q : p;
    }
    if (l15 == 0) atomicMin(&best[m], p);
  }
}

__global__ void __launch_bounds__(256)
init_best(unsigned long long* b, int n) {
  int i = blockIdx.x * 256 + threadIdx.x;
  if (i < n) b[i] = ~0ULL;
}

// gather z_q = codebook[idx], per-row commitment loss, write idx out
__global__ void __launch_bounds__(256)
vq_gather(const unsigned long long* __restrict__ best,
          const float* __restrict__ Z, const float* __restrict__ CB,
          float* __restrict__ zq, float* __restrict__ rowloss,
          int* __restrict__ idx_out) {
  __shared__ float red[256];
  int m = blockIdx.x, t = threadIdx.x;
  int idx = (int)(best[m] & 0xffffffffULL);
  float c = CB[(long long)idx * 256 + t];
  float d = c - Z[(long long)m * 256 + t];
  zq[(long long)m * 256 + t] = c;
  red[t] = d * d; __syncthreads();
  for (int o = 128; o > 0; o >>= 1) { if (t < o) red[t] += red[t + o]; __syncthreads(); }
  if (t == 0) { rowloss[m] = red[0]; idx_out[m] = idx; }
}

__global__ void __launch_bounds__(256)
loss_final(const float* __restrict__ rowloss, float* __restrict__ out) {
  __shared__ float red[256];
  int t = threadIdx.x;
  float s = 0.f;
  for (int i = t; i < 2048; i += 256) s += rowloss[i];
  red[t] = s; __syncthreads();
  for (int o = 128; o > 0; o >>= 1) { if (t < o) red[t] += red[t + o]; __syncthreads(); }
  if (t == 0) out[0] = 0.5f * red[0] / (2048.0f * 256.0f);
}

// ---------------- patchify / unpatchify ----------------
__global__ void __launch_bounds__(256)
patchify_k(const float* __restrict__ x, float* __restrict__ p) {
  long long i = (long long)blockIdx.x * 256 + threadIdx.x;  // over 8*256*3072
  int col = (int)(i % 3072);
  long long row = i / 3072;
  int n = (int)(row % 256);
  int b = (int)(row / 256);
  int c = col >> 8, rem = col & 255;
  int ph = rem >> 4, pw = rem & 15;
  int gh = n >> 4,  gw = n & 15;
  p[i] = x[(((long long)(b * 3 + c) * 256 + gh * 16 + ph) * 256) + gw * 16 + pw];
}

__global__ void __launch_bounds__(256)
unpatch_clamp(const float* __restrict__ om, float* __restrict__ out) {
  long long i = (long long)blockIdx.x * 256 + threadIdx.x;  // over 8*3*256*256
  int w = (int)(i & 255);
  long long t = i >> 8;
  int h = (int)(t & 255); t >>= 8;
  int c = (int)(t % 3);
  int b = (int)(t / 3);
  int gh = h >> 4, ph = h & 15, gw = w >> 4, pw = w & 15;
  int n = gh * 16 + gw;
  int col = c * 256 + ph * 16 + pw;
  float v = om[((long long)(b * 256 + n)) * 768 + col];
  out[i] = fminf(fmaxf(v, -1.0f), 1.0f);
}

// ---------------- host-side helpers ----------------
static void gemm(hipStream_t st,
                 const float* A, long long asm_, long long asb, long long ash,
                 const float* B, long long bsk, long long bsn, long long bsb, long long bsh,
                 float* C, long long csm, long long csb, long long csh,
                 const float* bias, const float* res, long long rsm, int rmask,
                 int M, int N, int K, int Z, int nh, float alpha, int act) {
  dim3 blk(32, 4, 1);
  dim3 grd((unsigned)((N + 63) / 64), (unsigned)((M + 63) / 64), (unsigned)Z);
  gemm_bf16<<<grd, blk, 0, st>>>(A, asm_, asb, ash, B, bsk, bsn, bsb, bsh,
                                 C, csm, csb, csh, bias, res, rsm, rmask,
                                 M, N, K, nh, alpha, act);
}

struct BlkW {
  const float *ln1g, *ln1b, *qkvw, *qkvb, *projw, *projb;
  const float *ln2g, *ln2b, *fc1w, *fc1b, *fc2w, *fc2b;
};

static void run_block(hipStream_t st, float* h, float* y, float* qkv,
                      float* scores, float* attnout, float* ffbuf, const BlkW& w) {
  layernorm_k<<<2048, 256, 0, st>>>(h, w.ln1g, w.ln1b, y, 768);
  // qkv = y @ W_qkv + b
  gemm(st, y, 768, 0, 0, w.qkvw, 2304, 1, 0, 0, qkv, 2304, 0, 0,
       w.qkvb, nullptr, 0, 0, 2048, 2304, 768, 1, 1, 1.0f, 0);
  // scores[b,h] = (Q @ K^T) * 1/sqrt(64)   (B = K^T: b_sk=1, b_sn=2304 -> LDS path)
  gemm(st, qkv, 2304, (long long)256 * 2304, 64,
       qkv + 768, 1, 2304, (long long)256 * 2304, 64,
       scores, 256, (long long)12 * 65536, 65536,
       nullptr, nullptr, 0, 0, 256, 256, 64, 96, 12, 0.125f, 0);
  softmax256<<<dim3(24576 / 8), dim3(32, 8), 0, st>>>(scores, 24576);
  // attnout[b,n,h*64+d] = att @ V
  gemm(st, scores, 256, (long long)12 * 65536, 65536,
       qkv + 1536, 2304, 1, (long long)256 * 2304, 64,
       attnout, 768, (long long)256 * 768, 64,
       nullptr, nullptr, 0, 0, 256, 64, 256, 96, 12, 1.0f, 0);
  // h = h + attnout @ W_proj + b
  gemm(st, attnout, 768, 0, 0, w.projw, 768, 1, 0, 0, h, 768, 0, 0,
       w.projb, h, 768, 2047, 2048, 768, 768, 1, 1, 1.0f, 0);
  layernorm_k<<<2048, 256, 0, st>>>(h, w.ln2g, w.ln2b, y, 768);
  // ffbuf = gelu(y @ fc1 + b)
  gemm(st, y, 768, 0, 0, w.fc1w, 3072, 1, 0, 0, ffbuf, 3072, 0, 0,
       w.fc1b, nullptr, 0, 0, 2048, 3072, 768, 1, 1, 1.0f, 1);
  // h = h + ffbuf @ fc2 + b
  gemm(st, ffbuf, 3072, 0, 0, w.fc2w, 768, 1, 0, 0, h, 768, 0, 0,
       w.fc2b, h, 768, 2047, 2048, 768, 3072, 1, 1, 1.0f, 0);
}

extern "C" void kernel_launch(void* const* d_in, const int* in_sizes, int n_in,
                              void* d_out, int out_size, void* d_ws, size_t ws_size,
                              hipStream_t stream) {
  (void)in_sizes; (void)n_in; (void)out_size; (void)ws_size;
  const float* x            = (const float*)d_in[0];
  const float* patch_w      = (const float*)d_in[1];
  const float* patch_b      = (const float*)d_in[2];
  const float* pos_enc      = (const float*)d_in[3];
  const float* pos_dec      = (const float*)d_in[4];
  const float* pre_quant_w  = (const float*)d_in[5];
  const float* pre_quant_b  = (const float*)d_in[6];
  const float* codebook     = (const float*)d_in[7];
  const float* post_quant_w = (const float*)d_in[8];
  const float* post_quant_b = (const float*)d_in[9];
  const float* dec_lnf_g    = (const float*)d_in[10];
  const float* dec_lnf_b    = (const float*)d_in[11];
  const float* unpatch_w    = (const float*)d_in[12];
  const float* unpatch_b    = (const float*)d_in[13];

  // workspace carve-out (floats)
  float* wf = (float*)d_ws;
  float* patches = wf; wf += 6291456;                 // [2048,3072]
  float* h       = wf; wf += 1572864;                 // [2048,768]
  float* y       = wf; wf += 1572864;                 // [2048,768]
  float* qkv     = wf; wf += 4718592;                 // [2048,2304]
  float* scores  = wf; wf += 6291456;                 // [96,256,256]
  float* attnout = wf; wf += 1572864;                 // [2048,768]
  float* ffbuf   = wf; wf += 6291456;                 // [2048,3072]
  float* z       = wf; wf += 524288;                  // [2048,256]
  float* zq      = wf; wf += 524288;                  // [2048,256]
  float* zn      = wf; wf += 2048;
  float* cn      = wf; wf += 8192;
  float* rowloss = wf; wf += 2048;
  unsigned long long* best = (unsigned long long*)wf; wf += 4096;
  float* outmat  = attnout;                           // reuse at the end

  auto mkblk = [&](int base, int l) -> BlkW {
    BlkW w;
    w.ln1g  = (const float*)d_in[base + 0]  + (long long)l * 768;
    w.ln1b  = (const float*)d_in[base + 1]  + (long long)l * 768;
    w.qkvw  = (const float*)d_in[base + 2]  + (long long)l * 768 * 2304;
    w.qkvb  = (const float*)d_in[base + 3]  + (long long)l * 2304;
    w.projw = (const float*)d_in[base + 4]  + (long long)l * 768 * 768;
    w.projb = (const float*)d_in[base + 5]  + (long long)l * 768;
    w.ln2g  = (const float*)d_in[base + 6]  + (long long)l * 768;
    w.ln2b  = (const float*)d_in[base + 7]  + (long long)l * 768;
    w.fc1w  = (const float*)d_in[base + 8]  + (long long)l * 768 * 3072;
    w.fc1b  = (const float*)d_in[base + 9]  + (long long)l * 3072;
    w.fc2w  = (const float*)d_in[base + 10] + (long long)l * 3072 * 768;
    w.fc2b  = (const float*)d_in[base + 11] + (long long)l * 768;
    return w;
  };

  // --- encode ---
  patchify_k<<<24576, 256, 0, stream>>>(x, patches);
  gemm(stream, patches, 3072, 0, 0, patch_w, 768, 1, 0, 0, h, 768, 0, 0,
       patch_b, pos_enc, 768, 255, 2048, 768, 3072, 1, 1, 1.0f, 0);
  for (int l = 0; l < 6; ++l)
    run_block(stream, h, y, qkv, scores, attnout, ffbuf, mkblk(14, l));

  // --- pre-quant ---
  gemm(stream, h, 768, 0, 0, pre_quant_w, 256, 1, 0, 0, z, 256, 0, 0,
       pre_quant_b, nullptr, 0, 0, 2048, 256, 768, 1, 1, 1.0f, 0);

  // --- VQ ---
  rownorm256<<<2048, 256, 0, stream>>>(z, zn);
  rownorm256<<<8192, 256, 0, stream>>>(codebook, cn);
  init_best<<<8, 256, 0, stream>>>(best, 2048);
  vq_argmin<<<dim3(8192 / 64, 2048 / 16), dim3(32, 4), 0, stream>>>(
      z, codebook, zn, cn, best, 256);
  int*   idx_out  = (int*)((float*)d_out + 1572864);
  float* loss_out = (float*)d_out + 1572864 + 2048;
  vq_gather<<<2048, 256, 0, stream>>>(best, z, codebook, zq, rowloss, idx_out);
  loss_final<<<1, 256, 0, stream>>>(rowloss, loss_out);

  // --- decode ---
  gemm(stream, zq, 256, 0, 0, post_quant_w, 768, 1, 0, 0, h, 768, 0, 0,
       post_quant_b, pos_dec, 768, 255, 2048, 768, 256, 1, 1, 1.0f, 0);
  for (int l = 0; l < 6; ++l)
    run_block(stream, h, y, qkv, scores, attnout, ffbuf, mkblk(26, l));

  layernorm_k<<<2048, 256, 0, stream>>>(h, dec_lnf_g, dec_lnf_b, y, 768);
  gemm(stream, y, 768, 0, 0, unpatch_w, 768, 1, 0, 0, outmat, 768, 0, 0,
       unpatch_b, nullptr, 0, 0, 2048, 768, 768, 1, 1, 1.0f, 0);
  unpatch_clamp<<<6144, 256, 0, stream>>>(outmat, (float*)d_out);
}
